// CrossImageContrastiveLoss_24077586661706
// MI455X (gfx1250) — matrix operand order repaired and checked by reading, not compiled, not run on hardware
//
#include <hip/hip_runtime.h>

typedef __attribute__((ext_vector_type(16))) _Float16 v16h;
typedef __attribute__((ext_vector_type(8)))  float    v8f;

#define N_PIX      4096     // 64*64 pixels
#define D_FEAT     64       // feature dim
#define NTILE      256      // N_PIX / 16
#define TAU_F      0.07f
#define EPS_F      1e-4f
#define NUM_LABELS 19
#define QCHUNKS    4
#define QT_PER_CHUNK (NTILE / QCHUNKS)
// exp(x/tau) == exp2(x * LOG2E_OVER_TAU)  -> single mul + native v_exp_f32
#define LOG2E_OVER_TAU 20.609929155556627f

// ---------------------------------------------------------------------------
// Pack f32 [D, N] feature arrays into f16 WMMA fragment order.
// A fragment (16x32, MxK): lanes 0-15 -> M=lane, halves i = K i;
// lanes 16-31 -> M=lane-16, halves i = K 16+i.
// B fragment (32x16, KxN): lanes 0-15 -> N=lane, halves i = K i;
// lanes 16-31 -> N=lane-16, halves i = K 16+i.
// One thread per (tile, kstep, lane); each writes 16 contiguous halves (32 B).
// ---------------------------------------------------------------------------
__global__ __launch_bounds__(256) void pack_frags(
    const float* __restrict__ Fi, const float* __restrict__ Fii,
    const float* __restrict__ Fjj,
    _Float16* __restrict__ Apack, _Float16* __restrict__ BiiP,
    _Float16* __restrict__ BjjP) {
  int t = blockIdx.x * blockDim.x + threadIdx.x;  // (tile*2 + step)*32 + lane
  int lane = t & 31;
  int step = (t >> 5) & 1;
  int tile = t >> 6;
  if (tile >= NTILE) return;
  int col = lane & 15;

  // ---- A fragment from Fi ----
  {
    int p = tile * 16 + col;                       // pixel row m
    int kbase = step * 32 + ((lane < 16) ? 0 : 16);
    _Float16* dst = Apack + (size_t)t * 16;
#pragma unroll
    for (int i = 0; i < 16; ++i) {
      int k = kbase + i;
      dst[i] = (_Float16)Fi[k * N_PIX + p];
    }
  }
  // ---- B fragments from Fii / Fjj ----
  {
    int q = tile * 16 + col;                       // pixel column n
    int kbase = step * 32 + ((lane < 16) ? 0 : 16);
    _Float16* d1 = BiiP + (size_t)t * 16;
    _Float16* d2 = BjjP + (size_t)t * 16;
#pragma unroll
    for (int i = 0; i < 16; ++i) {
      int k = kbase + i;
      d1[i] = (_Float16)Fii[k * N_PIX + q];
      d2[i] = (_Float16)Fjj[k * N_PIX + q];
    }
  }
}

// ---------------------------------------------------------------------------
// Exact f32 diagonals of the two logit matrices: diag[p] = (Fi_p . F_p)/tau
// ---------------------------------------------------------------------------
__global__ __launch_bounds__(256) void diag_kernel(
    const float* __restrict__ Fi, const float* __restrict__ Fii,
    const float* __restrict__ Fjj, float* __restrict__ diag1,
    float* __restrict__ diag2) {
  int p = blockIdx.x * blockDim.x + threadIdx.x;
  if (p >= N_PIX) return;
  float a = 0.0f, b = 0.0f;
#pragma unroll 8
  for (int c = 0; c < D_FEAT; ++c) {
    float f = Fi[c * N_PIX + p];
    a += f * Fii[c * N_PIX + p];
    b += f * Fjj[c * N_PIX + p];
  }
  diag1[p] = a * (1.0f / TAU_F);
  diag2[p] = b * (1.0f / TAU_F);
}

// ---------------------------------------------------------------------------
// Fused GEMM + exp + masked row-reduction.
// Block mapping: the 4 waves of a block own 4 *different* p-tiles but walk the
// *same* q-stream (same qchunk) -> identical B-fragment cachelines hit in the
// WGP cache instead of 4x L2 traffic. One merged accumulator per row since
// only sum_s1 + diff_sum reaches memory.
// Zpart layout: [QCHUNKS][N_PIX]; each element written exactly once (no
// atomics -> deterministic).
// ---------------------------------------------------------------------------
__global__ __launch_bounds__(128) void main_kernel(
    const _Float16* __restrict__ Apack, const _Float16* __restrict__ BiiP,
    const _Float16* __restrict__ BjjP, const int* __restrict__ lab,
    const int* __restrict__ jj, float* __restrict__ Zpart) {
  int widx = threadIdx.x >> 5;
  int lane = threadIdx.x & 31;
  int qchunk = blockIdx.x & (QCHUNKS - 1);
  int ptile  = (blockIdx.x >> 2) * 4 + widx;   // 4 p-tiles share one q-stream
  int col = lane & 15;

  // A fragments for this p-tile: resident in registers for the whole loop.
  const v16h a0 = *(const v16h*)(Apack + ((size_t)(ptile * 2 + 0) * 32 + lane) * 16);
  const v16h a1 = *(const v16h*)(Apack + ((size_t)(ptile * 2 + 1) * 32 + lane) * 16);

  // Labels of the 8 matrix rows this lane's accumulator VGPRs cover.
  int rbase = ptile * 16 + ((lane < 16) ? 0 : 8);
  int labRow[8];
#pragma unroll
  for (int r = 0; r < 8; ++r) labRow[r] = lab[rbase + r];

  float zacc[8];
#pragma unroll
  for (int r = 0; r < 8; ++r) zacc[r] = 0.0f;

  int q0 = qchunk * QT_PER_CHUNK;

  for (int qt = q0; qt < q0 + QT_PER_CHUNK; ++qt) {
    const _Float16* bBase = BiiP + ((size_t)(qt * 2) * 32 + lane) * 16;
    const _Float16* cBase = BjjP + ((size_t)(qt * 2) * 32 + lane) * 16;
    const v16h b0 = *(const v16h*)(bBase);
    const v16h b1 = *(const v16h*)(bBase + 32 * 16);
    const v16h c0 = *(const v16h*)(cBase);
    const v16h c1 = *(const v16h*)(cBase + 32 * 16);

    // prefetch B fragments a few q-tiles ahead (global_prefetch_b8)
    if (qt + 4 < q0 + QT_PER_CHUNK) {
      __builtin_prefetch(bBase + (size_t)4 * 2 * 32 * 16, 0, 3);
      __builtin_prefetch(cBase + (size_t)4 * 2 * 32 * 16, 0, 3);
    }

    v8f acc1 = {};
    acc1 = __builtin_amdgcn_wmma_f32_16x16x32_f16(false, a0, false, b0,
                                                  (short)0, acc1, false, false);
    acc1 = __builtin_amdgcn_wmma_f32_16x16x32_f16(false, a1, false, b1,
                                                  (short)0, acc1, false, false);
    v8f acc2 = {};
    acc2 = __builtin_amdgcn_wmma_f32_16x16x32_f16(false, a0, false, c0,
                                                  (short)0, acc2, false, false);
    acc2 = __builtin_amdgcn_wmma_f32_16x16x32_f16(false, a1, false, c1,
                                                  (short)0, acc2, false, false);

    int jq = jj[qt * 16 + col];  // label of this lane's column q
#pragma unroll
    for (int r = 0; r < 8; ++r) {
      float e1 = __builtin_amdgcn_exp2f(acc1[r] * LOG2E_OVER_TAU);
      float e2 = __builtin_amdgcn_exp2f(acc2[r] * LOG2E_OVER_TAU);
      zacc[r] += e1 + ((jq == labRow[r]) ? e2 : 0.0f);
    }
  }

  // Reduce across the 16 lanes that share each matrix row (wave32, width 16).
#pragma unroll
  for (int r = 0; r < 8; ++r) {
    float v = zacc[r];
    v += __shfl_xor(v, 1, 16);
    v += __shfl_xor(v, 2, 16);
    v += __shfl_xor(v, 4, 16);
    v += __shfl_xor(v, 8, 16);
    if (col == 0) Zpart[qchunk * N_PIX + rbase + r] = v;
  }
}

// ---------------------------------------------------------------------------
// Final: label counts in LDS (integer-valued float adds -> deterministic),
// combine Z partials, per-pixel loss term, block tree-reduction -> scalar.
// ---------------------------------------------------------------------------
__global__ __launch_bounds__(256) void final_kernel(
    const int* __restrict__ lab, const int* __restrict__ ii,
    const int* __restrict__ jj, const float* __restrict__ diag1,
    const float* __restrict__ diag2, const float* __restrict__ Zpart,
    float* __restrict__ out) {
  __shared__ float cnt_ii_s[NUM_LABELS];
  __shared__ float cnt_jj_s[NUM_LABELS];
  __shared__ float red[256];
  int t = threadIdx.x;
  if (t < NUM_LABELS) { cnt_ii_s[t] = 0.0f; cnt_jj_s[t] = 0.0f; }
  __syncthreads();
  for (int p = t; p < N_PIX; p += 256) {
    atomicAdd(&cnt_ii_s[ii[p]], 1.0f);
    atomicAdd(&cnt_jj_s[jj[p]], 1.0f);
  }
  __syncthreads();

  float acc = 0.0f;
  for (int p = t; p < N_PIX; p += 256) {
    float Z = Zpart[p] + Zpart[N_PIX + p] + Zpart[2 * N_PIX + p] +
              Zpart[3 * N_PIX + p] + EPS_F;
    float lz = __logf(Z);
    int L = lab[p];
    float ci = cnt_ii_s[L];
    float w = ci / (ci + cnt_jj_s[L] + EPS_F);
    acc += w * (diag1[p] + diag2[p] - 2.0f * lz);
  }
  red[t] = acc;
  __syncthreads();
#pragma unroll
  for (int s = 128; s > 0; s >>= 1) {
    if (t < s) red[t] += red[t + s];
    __syncthreads();
  }
  if (t == 0) out[0] = -red[0] / (float)N_PIX;
}

// ---------------------------------------------------------------------------
extern "C" void kernel_launch(void* const* d_in, const int* in_sizes, int n_in,
                              void* d_out, int out_size, void* d_ws,
                              size_t ws_size, hipStream_t stream) {
  const float* Fi  = (const float*)d_in[0];
  const float* Fii = (const float*)d_in[1];
  const float* Fjj = (const float*)d_in[2];
  const int* lab   = (const int*)d_in[3];  // labels "i"
  const int* iiL   = (const int*)d_in[4];  // labels "ii"
  const int* jjL   = (const int*)d_in[5];  // labels "jj"

  // Workspace layout (~1.6 MB total):
  _Float16* Apack = (_Float16*)d_ws;                 // 4096*64 halves (512 KB)
  _Float16* BiiP  = Apack + (size_t)N_PIX * D_FEAT;  // 512 KB
  _Float16* BjjP  = BiiP + (size_t)N_PIX * D_FEAT;   // 512 KB
  float* diag1 = (float*)(BjjP + (size_t)N_PIX * D_FEAT);  // 16 KB
  float* diag2 = diag1 + N_PIX;                            // 16 KB
  float* Zpart = diag2 + N_PIX;                            // QCHUNKS*N (64 KB)

  // 1) repack features to f16 WMMA fragment order
  pack_frags<<<(NTILE * 2 * 32) / 256, 256, 0, stream>>>(Fi, Fii, Fjj, Apack,
                                                         BiiP, BjjP);
  // 2) exact diagonals in f32
  diag_kernel<<<N_PIX / 256, 256, 0, stream>>>(Fi, Fii, Fjj, diag1, diag2);
  // 3) fused WMMA GEMM + exp + masked row reduction
  //    1024 waves = 256 blocks x 4 waves (128 threads, wave32)
  main_kernel<<<NTILE, 128, 0, stream>>>(Apack, BiiP, BjjP, lab, jjL, Zpart);
  // 4) scalar loss
  final_kernel<<<1, 256, 0, stream>>>(lab, iiL, jjL, diag1, diag2, Zpart,
                                      (float*)d_out);
}
